// SimpleRNN_61031485276656
// MI455X (gfx1250) — matrix operand order repaired and checked by reading, not compiled, not run on hardware
//
#include <hip/hip_runtime.h>
#include <math.h>

// ---------------------------------------------------------------------------
// SimpleRNN: B=256, T=512, I=64, H=512, O=1
// 16 workgroups x 16 batch rows; 512 threads = 16 waves x 2 N-tiles.
// Hidden state 16x512 bf16 double-buffered in LDS; recurrent + input GEMMs
// via v_wmma_f32_16x16x32_bf16 on pre-swizzled fragment-native operands.
// A laundered per-step integer offset defeats LICM (no preload/spill), and
// explicit addrspace(1) loads guarantee global_load_b128 (not flat_load).
// ---------------------------------------------------------------------------

typedef __attribute__((ext_vector_type(16))) __bf16        v16bf;
typedef __attribute__((ext_vector_type(8)))  float         v8f;
typedef __attribute__((ext_vector_type(4)))  unsigned int  u32x4;
typedef __attribute__((address_space(1))) const u32x4      gu32x4;

union frag16 { u32x4 q[2]; v16bf v; };

#define Bn 256
#define Tn 512
#define In 64
#define Hn 512
#define HOn 256
#define BLK 512   // threads per workgroup (16 waves)

// ws layout (bytes)
#define WHH_OFF   0u           // 32 ntiles * 16 kblocks * 1024 B = 512 KB
#define WIH_OFF   524288u      // 32 ntiles *  2 kblocks * 1024 B = 64 KB
#define BIAS_OFF  589824u      // 512 f32 = 2 KB
#define XF_OFF    591872u      // 16 btiles * 512 t * 2 kblocks * 1024 B = 16 MB

__device__ __forceinline__ unsigned int pack_bf2(float a, float b) {
  unsigned int ua = __float_as_uint(a), ub = __float_as_uint(b);
  unsigned int ra = (ua + 0x7FFFu + ((ua >> 16) & 1u)) >> 16;
  unsigned int rb = (ub + 0x7FFFu + ((ub >> 16) & 1u)) >> 16;
  return (ra & 0xFFFFu) | (rb << 16);
}
__device__ __forceinline__ unsigned short f2bf(float a) {
  unsigned int ua = __float_as_uint(a);
  return (unsigned short)((ua + 0x7FFFu + ((ua >> 16) & 1u)) >> 16);
}
__device__ __forceinline__ float bf2f(unsigned short h) {
  return __uint_as_float(((unsigned int)h) << 16);
}

// Force a 32-byte fragment load down the GLOBAL path (global_load_b128).
__device__ __forceinline__ void load_frag_g(frag16& f, const unsigned int* p) {
  const gu32x4* q = (const gu32x4*)(const void*)p;  // addrspacecast flat->global
  f.q[0] = q[0];
  f.q[1] = q[1];
}

// Pre-swizzle row-major W[N x K] (used as B = W^T) into B-fragment layout:
// tile (nt,kt): 256 uint32, index = lane*8 + v
// lane: n_local = lane&15, khalf = lane>>4 ; VGPR v holds K = khalf*16+2v, +1
__global__ void prep_wfrag(const float* __restrict__ W, int K, int Kblocks,
                           unsigned int* __restrict__ dst, int total) {
  int tid = blockIdx.x * blockDim.x + threadIdx.x;
  if (tid >= total) return;
  int r    = tid & 255;
  int frag = tid >> 8;
  int kt = frag % Kblocks;
  int nt = frag / Kblocks;
  int lane = r >> 3, v = r & 7;
  int n  = nt * 16 + (lane & 15);
  int k0 = kt * 32 + (lane >> 4) * 16 + v * 2;
  size_t base = (size_t)n * K + k0;
  dst[tid] = pack_bf2(W[base], W[base + 1]);
}

// Pre-swizzle x[B,T,I] into A-fragment layout per (btile, t, kblock):
// lane: m = lane&15, half = lane>>4
// VGPR v: K = kb*32 + half*16 + v*2, +1
__global__ void prep_xfrag(const float* __restrict__ x,
                           unsigned int* __restrict__ dst) {
  int tid  = blockIdx.x * blockDim.x + threadIdx.x; // 16*512*2*256 threads
  int r    = tid & 255;
  int frag = tid >> 8;
  int kb    = frag & 1;
  int bt_t  = frag >> 1;
  int t     = bt_t & 511;
  int btile = bt_t >> 9;
  int lane = r >> 3, v = r & 7;
  int m = lane & 15, half = lane >> 4;
  int b  = btile * 16 + m;
  int k0 = kb * 32 + half * 16 + v * 2;
  size_t base = ((size_t)b * Tn + t) * In + k0;
  dst[tid] = pack_bf2(x[base], x[base + 1]);
}

__global__ void prep_bias(const float* __restrict__ a,
                          const float* __restrict__ b,
                          float* __restrict__ dst) {
  int i = blockIdx.x * blockDim.x + threadIdx.x;
  if (i < Hn) dst[i] = a[i] + b[i];
}

// Main RNN scan + FC head.
__global__ __launch_bounds__(BLK) void rnn_scan_kernel(
    const unsigned int* __restrict__ whh_f,
    const unsigned int* __restrict__ wih_f,
    const unsigned int* __restrict__ x_f,
    const float* __restrict__ bias,
    const float* __restrict__ W1, const float* __restrict__ b1,
    const float* __restrict__ W2, const float* __restrict__ b2,
    float* __restrict__ out) {
  extern __shared__ unsigned char smem[];
  unsigned short* h0 = (unsigned short*)smem;          // 16x512 bf16 (16 KB)
  unsigned short* h1 = h0 + 16 * Hn;                   // 16x512 bf16 (16 KB)
  float* z = (float*)(smem + 16 * Hn * 2);             // 16x256 f32, reuses h1

  const int tid   = threadIdx.x;
  const int lane  = tid & 31;
  const int wave  = tid >> 5;          // 0..15
  const int wg    = blockIdx.x;
  const int lrow  = lane & 15;
  const int lhalf = lane >> 4;

  // h(0) = 0
  for (int i = tid; i < 16 * Hn / 2; i += BLK) ((unsigned int*)h0)[i] = 0u;

  // bias fragment per owned N-tile (C layout: column = lane&15)
  float bfr[2];
#pragma unroll
  for (int j = 0; j < 2; ++j) bfr[j] = bias[(wave * 2 + j) * 16 + lrow];

  const unsigned int* xf = x_f + (size_t)wg * Tn * 2 * 256;

  __syncthreads();

  for (int t = 0; t < Tn; ++t) {
    const unsigned short* hr = (t & 1) ? h1 : h0;
    unsigned short*       hw = (t & 1) ? h0 : h1;

    // Loop-variant opaque offset: keeps the weight-fragment loads inside the
    // t-loop (no LICM -> no register-file preload/spill), while the pointer
    // chain stays rooted at the kernel args (global address space).
    unsigned int toff = 0;
    asm volatile("" : "+s"(toff));
    const unsigned int* whh_t = whh_f + toff;
    const unsigned int* wih_t = wih_f + toff;

#pragma unroll
    for (int j = 0; j < 2; ++j) {
      const int nt = wave * 2 + j;
      const float bv = bfr[j];
      v8f acc = {bv, bv, bv, bv, bv, bv, bv, bv};

      // input projection: K = 64 (2 kblocks)
#pragma unroll
      for (int kb = 0; kb < 2; ++kb) {
        frag16 a, b;
        load_frag_g(a, xf + (t * 2 + kb) * 256 + lane * 8);
        load_frag_g(b, wih_t + (nt * 2 + kb) * 256 + lane * 8);
        acc = __builtin_amdgcn_wmma_f32_16x16x32_bf16(
            false, a.v, false, b.v, (short)0, acc, false, false);
      }

      // recurrence: K = 512 (16 kblocks); A-fragments direct from LDS
#pragma unroll
      for (int kb = 0; kb < 16; ++kb) {
        frag16 a, b;
        const unsigned short* hp = hr + lrow * Hn + kb * 32 + lhalf * 16;
        a.q[0] = *(const u32x4*)(hp);
        a.q[1] = *(const u32x4*)(hp + 8);
        load_frag_g(b, whh_t + (nt * 16 + kb) * 256 + lane * 8);
        acc = __builtin_amdgcn_wmma_f32_16x16x32_bf16(
            false, a.v, false, b.v, (short)0, acc, false, false);
      }

      // tanh + writeback: D layout lane -> (M = v + 8*half, N = lane&15)
      const int col = nt * 16 + lrow;
#pragma unroll
      for (int v = 0; v < 8; ++v) {
        int row = v + 8 * lhalf;
        hw[row * Hn + col] = f2bf(tanhf(acc[v]));
      }
    }
    __syncthreads();  // next step reads what all waves just wrote
  }

  // h_last is in h0 (T even). FC head.
  for (int idx = tid; idx < 16 * HOn; idx += BLK) {
    int m = idx >> 8, jj = idx & 255;
    float acc = b1[jj];
    const float* w = W1 + (size_t)jj * Hn;
    for (int k = 0; k < Hn; ++k) acc += bf2f(h0[m * Hn + k]) * w[k];
    z[m * HOn + jj] = acc > 0.f ? acc : 0.f;
  }
  __syncthreads();
  if (tid < 16) {
    float acc = b2[0];
    for (int jj = 0; jj < HOn; ++jj) acc += z[tid * HOn + jj] * W2[jj];
    out[wg * 16 + tid] = 1.f / (1.f + expf(-acc));
  }
}

extern "C" void kernel_launch(void* const* d_in, const int* in_sizes, int n_in,
                              void* d_out, int out_size, void* d_ws, size_t ws_size,
                              hipStream_t stream) {
  (void)in_sizes; (void)n_in; (void)out_size; (void)ws_size;
  const float* x   = (const float*)d_in[0];
  const float* Wih = (const float*)d_in[1];
  const float* Whh = (const float*)d_in[2];
  const float* bih = (const float*)d_in[3];
  const float* bhh = (const float*)d_in[4];
  const float* W1  = (const float*)d_in[5];
  const float* b1  = (const float*)d_in[6];
  const float* W2  = (const float*)d_in[7];
  const float* b2  = (const float*)d_in[8];
  float* out = (float*)d_out;

  unsigned char* ws = (unsigned char*)d_ws;
  unsigned int* whh_f = (unsigned int*)(ws + WHH_OFF);
  unsigned int* wih_f = (unsigned int*)(ws + WIH_OFF);
  float*        bias  = (float*)(ws + BIAS_OFF);
  unsigned int* x_f   = (unsigned int*)(ws + XF_OFF);

  prep_wfrag<<<512, 256, 0, stream>>>(Whh, Hn, 16, whh_f, 32 * 16 * 256);
  prep_wfrag<<<64, 256, 0, stream>>>(Wih, In, 2, wih_f, 32 * 2 * 256);
  prep_bias<<<2, 256, 0, stream>>>(bih, bhh, bias);
  prep_xfrag<<<16384, 256, 0, stream>>>(x, x_f);

  rnn_scan_kernel<<<16, BLK, 32768, stream>>>(whh_f, wih_f, x_f, bias,
                                              W1, b1, W2, b2, out);
}
